// CompositeTransitionNet_77506979824204
// MI455X (gfx1250) — compile-verified, matrix operand order checked by reading, not run on hardware
//
#include <hip/hip_runtime.h>
#include <hip/hip_bf16.h>
#include <math.h>

// ---------------------------------------------------------------------------
// CDNA5 / gfx1250 implementation.
// Heavy GEMMs on v_wmma_f32_16x16x32_bf16 (f32 staged->bf16 in LDS, f32 acc).
// bf16-A GEMMs stage their A tile with GLOBAL_LOAD_ASYNC_TO_LDS_B128.
// wave32, 8 waves / 256-thread block.
// ---------------------------------------------------------------------------

typedef __attribute__((ext_vector_type(16))) __bf16 v16bf;
typedef __attribute__((ext_vector_type(8)))  __bf16 v8bf;
typedef __attribute__((ext_vector_type(8)))  float  v8f;
typedef int v4i __attribute__((vector_size(16)));   // matches async-LDS builtin

#define B_  4096
#define D_  1024
#define H_  4096
#define E_  8
#define K_  4096
#define V_  50257

// Native bf16 convert (FPTrunc, round-to-nearest-even on gfx1250).
__device__ __forceinline__ __bf16 f2bf(float f) { return (__bf16)f; }

__device__ __forceinline__ v16bf comb16(v8bf lo, v8bf hi) {
  v16bf o;
#pragma unroll
  for (int i = 0; i < 8; ++i) { o[i] = lo[i]; o[i + 8] = hi[i]; }
  return o;
}

__device__ __forceinline__ void wait_asynccnt0() {
#if __has_builtin(__builtin_amdgcn_s_wait_asynccnt)
  __builtin_amdgcn_s_wait_asynccnt(0);
#else
  asm volatile("s_wait_asynccnt 0x0" ::: "memory");
#endif
}

// Async 16B global->LDS copy (ASYNCcnt-tracked).
__device__ __forceinline__ void async_copy_b128(const void* gsrc, void* ldst) {
#if __has_builtin(__builtin_amdgcn_global_load_async_to_lds_b128)
  typedef __attribute__((address_space(1))) v4i* gp_t;
  typedef __attribute__((address_space(3))) v4i* lp_t;
  __builtin_amdgcn_global_load_async_to_lds_b128(
      (gp_t)(void*)gsrc, (lp_t)ldst, 0, 0);
#else
  unsigned lds_addr = (unsigned)(size_t)ldst;
  unsigned long long ga = (unsigned long long)(size_t)gsrc;
  asm volatile("global_load_async_to_lds_b128 %0, %1, off"
               :: "v"(lds_addr), "v"(ga) : "memory");
#endif
}

enum {
  EPI_STORE_F32 = 0,    // out = acc + bias
  EPI_ADD_F32,          // out += acc
  EPI_RELU_BF16,        // out(bf16) = relu(acc + bias)
  EPI_GELU_BF16,        // out(bf16) = gelu_exact(acc + bias)
  EPI_SCALED_ACC_F32,   // out += scale[row]*(acc + bias)
  EPI_GATE_F32          // out = prev + sigmoid(acc + bias) * aux
};

// Tile geometry
#define BM 64
#define BN 128
#define BK 32
#define LDK 40   // padded bf16 row (80B): 16B aligned, bank-conflict-free frags

// C = act( A[M,Kd] @ W^T + bias ), W stored [N,Kd] (torch Linear) or,
// when BT=true, stored [Kd,N] (e.g. `values`).
template <typename AT, bool BT, int EPI>
__global__ void __launch_bounds__(256)
wmma_gemm(const AT* __restrict__ A, int lda,
          const float* __restrict__ W, int ldb,
          const float* __restrict__ bias,
          const float* __restrict__ scale, int scale_stride,
          const float* __restrict__ prev,
          const float* __restrict__ aux,
          void* __restrict__ Out, int ldo,
          int Kd, int Nsize) {
  __shared__ __align__(16) __bf16 sA[2][BM][LDK];
  __shared__ __align__(16) __bf16 sB[2][BN][LDK];

  const int tid = threadIdx.x;
  const int m0  = blockIdx.y * BM;
  const int n0  = blockIdx.x * BN;

  auto stage = [&](int buf, int k0) {
    // ---- A tile: 64 x 32, 16B per thread ----
    {
      int row = tid >> 2;
      int kb  = (tid & 3) * 8;
      const AT* src = A + (size_t)(m0 + row) * lda + k0 + kb;
      if constexpr (__is_same(AT, float)) {
        const float4* p = reinterpret_cast<const float4*>(src);
        float4 x = p[0], y = p[1];
        v8bf v = {f2bf(x.x), f2bf(x.y), f2bf(x.z), f2bf(x.w),
                  f2bf(y.x), f2bf(y.y), f2bf(y.z), f2bf(y.w)};
        *reinterpret_cast<v8bf*>(&sA[buf][row][kb]) = v;
      } else {
        // A already bf16: async DMA global -> LDS, 16B/lane, ASYNCcnt-tracked.
        async_copy_b128(src, &sA[buf][row][kb]);
      }
    }
    // ---- W tile: 128 x 32 (always f32 in global) ----
    if constexpr (!BT) {
#pragma unroll
      for (int j = 0; j < 2; ++j) {
        int row = (tid >> 2) + j * 64;
        int kb  = (tid & 3) * 8;
        v8bf v = {};
        if (n0 + row < Nsize) {
          const float4* p = reinterpret_cast<const float4*>(
              W + (size_t)(n0 + row) * ldb + k0 + kb);
          float4 x = p[0], y = p[1];
          v = v8bf{f2bf(x.x), f2bf(x.y), f2bf(x.z), f2bf(x.w),
                   f2bf(y.x), f2bf(y.y), f2bf(y.z), f2bf(y.w)};
          __builtin_prefetch(W + (size_t)(n0 + row) * ldb + k0 + kb + 2 * BK, 0, 1);
        }
        *reinterpret_cast<v8bf*>(&sB[buf][row][kb]) = v;
      }
    } else {
      // W stored [Kd, N]: gather column-major into row-major LDS tile.
#pragma unroll
      for (int it = 0; it < (BN * BK) / 256; ++it) {
        int idx = tid + it * 256;
        int k = idx >> 7;        // 0..31
        int n = idx & 127;       // 0..127
        float v = (n0 + n < Nsize) ? W[(size_t)(k0 + k) * ldb + n0 + n] : 0.f;
        sB[buf][n][k] = f2bf(v);
      }
    }
  };

  const int wave = tid >> 5, lane = tid & 31;
  const int wm = (wave >> 2) * 32;          // 0 / 32
  const int wn = (wave & 3) * 32;           // 0 / 32 / 64 / 96
  const int half = lane >> 4, r = lane & 15;
  const int c0 = half * 8, c1 = 16 + half * 8;   // ISA 16-bit A/B K-chunks

  v8f acc[2][2] = {};
  const int nk = Kd / BK;

  stage(0, 0);
  if constexpr (__is_same(AT, __bf16)) wait_asynccnt0();
  __syncthreads();

  for (int kt = 0; kt < nk; ++kt) {
    const int buf = kt & 1;
    if (kt + 1 < nk) stage(buf ^ 1, (kt + 1) * BK);

    v16bf af[2], bf[2];
#pragma unroll
    for (int i = 0; i < 2; ++i) {
      const __bf16* ap = &sA[buf][wm + i * 16 + r][0];
      af[i] = comb16(*(const v8bf*)(ap + c0), *(const v8bf*)(ap + c1));
      const __bf16* bp = &sB[buf][wn + i * 16 + r][0];
      bf[i] = comb16(*(const v8bf*)(bp + c0), *(const v8bf*)(bp + c1));
    }
#pragma unroll
    for (int i = 0; i < 2; ++i)
#pragma unroll
      for (int j = 0; j < 2; ++j)
        acc[i][j] = __builtin_amdgcn_wmma_f32_16x16x32_bf16(
            false, af[i], false, bf[j], (short)0, acc[i][j], false, false);

    if constexpr (__is_same(AT, __bf16)) wait_asynccnt0();
    __syncthreads();
  }

  // ---- epilogue (C layout: elem t -> M = t + 8*half, N = lane&15) ----
#pragma unroll
  for (int i = 0; i < 2; ++i)
#pragma unroll
    for (int j = 0; j < 2; ++j) {
      const int col = n0 + wn + j * 16 + (lane & 15);
      if (col >= Nsize) continue;
      const float bb = bias ? bias[col] : 0.f;
      const int rowbase = m0 + wm + i * 16 + half * 8;
#pragma unroll
      for (int t = 0; t < 8; ++t) {
        const int row = rowbase + t;
        float v = acc[i][j][t] + bb;
        const size_t o = (size_t)row * ldo + col;
        if constexpr (EPI == EPI_STORE_F32) {
          ((float*)Out)[o] = v;
        } else if constexpr (EPI == EPI_ADD_F32) {
          ((float*)Out)[o] += v;
        } else if constexpr (EPI == EPI_RELU_BF16) {
          ((__bf16*)Out)[o] = f2bf(fmaxf(v, 0.f));
        } else if constexpr (EPI == EPI_GELU_BF16) {
          ((__bf16*)Out)[o] = f2bf(0.5f * v * (1.f + erff(v * 0.70710678118f)));
        } else if constexpr (EPI == EPI_SCALED_ACC_F32) {
          ((float*)Out)[o] += scale[(size_t)row * scale_stride] * v;
        } else if constexpr (EPI == EPI_GATE_F32) {
          float g = 1.f / (1.f + __expf(-v));
          ((float*)Out)[o] = prev[o] + g * aux[o];
        }
      }
    }
}

// ---- gate GEMM (B x 1024 x 8) + softmax + top-2 -> dense sparse weights ----
__global__ void __launch_bounds__(256)
gate_topk_kernel(const float* __restrict__ z, const float* __restrict__ gw,
                 const float* __restrict__ gb, float* __restrict__ we) {
  const int wave = threadIdx.x >> 5, lane = threadIdx.x & 31;
  const int row = blockIdx.x * 8 + wave;
  const float* zr = z + (size_t)row * D_;
  float acc[E_] = {};
  for (int d = lane; d < D_; d += 32) {
    float zv = zr[d];
#pragma unroll
    for (int e = 0; e < E_; ++e) acc[e] += zv * gw[e * D_ + d];
  }
#pragma unroll
  for (int e = 0; e < E_; ++e) {
    float v = acc[e];
#pragma unroll
    for (int off = 16; off > 0; off >>= 1) v += __shfl_xor(v, off, 32);
    acc[e] = v;
  }
  if (lane == 0) {
    float mx = -1e30f;
#pragma unroll
    for (int e = 0; e < E_; ++e) { acc[e] += gb[e]; mx = fmaxf(mx, acc[e]); }
    float p[E_], s = 0.f;
#pragma unroll
    for (int e = 0; e < E_; ++e) { p[e] = __expf(acc[e] - mx); s += p[e]; }
    float inv = 1.f / s;
#pragma unroll
    for (int e = 0; e < E_; ++e) p[e] *= inv;
    int i1 = 0;
#pragma unroll
    for (int e = 1; e < E_; ++e) if (p[e] > p[i1]) i1 = e;
    int i2 = (i1 == 0) ? 1 : 0;
#pragma unroll
    for (int e = 0; e < E_; ++e) if (e != i1 && p[e] > p[i2]) i2 = e;
#pragma unroll
    for (int e = 0; e < E_; ++e)
      we[(size_t)row * E_ + e] = (e == i1 || e == i2) ? p[e] : 0.f;
  }
}

// ---- row softmax over K=4096, f32 scores -> bf16 probabilities ----
__global__ void __launch_bounds__(256)
row_softmax_bf16(const float* __restrict__ S, __bf16* __restrict__ P, int K) {
  __shared__ float red[8];
  const int row = blockIdx.x, tid = threadIdx.x;
  const int lane = tid & 31, wv = tid >> 5;
  const float* s = S + (size_t)row * K;
  float mx = -1e30f;
  for (int k = tid; k < K; k += 256) mx = fmaxf(mx, s[k]);
#pragma unroll
  for (int off = 16; off > 0; off >>= 1) mx = fmaxf(mx, __shfl_xor(mx, off, 32));
  if (lane == 0) red[wv] = mx;
  __syncthreads();
  float bm = red[0];
#pragma unroll
  for (int i = 1; i < 8; ++i) bm = fmaxf(bm, red[i]);
  __syncthreads();
  float sum = 0.f;
  for (int k = tid; k < K; k += 256) sum += __expf(s[k] - bm);
#pragma unroll
  for (int off = 16; off > 0; off >>= 1) sum += __shfl_xor(sum, off, 32);
  if (lane == 0) red[wv] = sum;
  __syncthreads();
  float bs = 0.f;
#pragma unroll
  for (int i = 0; i < 8; ++i) bs += red[i];
  const float inv = 1.f / bs;
  for (int k = tid; k < K; k += 256)
    P[(size_t)row * K + k] = f2bf(__expf(s[k] - bm) * inv);
}

extern "C" void kernel_launch(void* const* d_in, const int* in_sizes, int n_in,
                              void* d_out, int out_size, void* d_ws, size_t ws_size,
                              hipStream_t stream) {
  const float* z      = (const float*)d_in[0];
  const float* gate_w = (const float*)d_in[1];
  const float* gate_b = (const float*)d_in[2];
  const float* ew1    = (const float*)d_in[3];
  const float* eb1    = (const float*)d_in[4];
  const float* ew2    = (const float*)d_in[5];
  const float* eb2    = (const float*)d_in[6];
  const float* keys   = (const float*)d_in[7];
  const float* values = (const float*)d_in[8];
  const float* m1f1w = (const float*)d_in[9],  *m1f1b = (const float*)d_in[10];
  const float* m1f2w = (const float*)d_in[11], *m1f2b = (const float*)d_in[12];
  const float* m1gw  = (const float*)d_in[13], *m1gb  = (const float*)d_in[14];
  const float* m2f1w = (const float*)d_in[15], *m2f1b = (const float*)d_in[16];
  const float* m2f2w = (const float*)d_in[17], *m2f2b = (const float*)d_in[18];
  const float* m2gw  = (const float*)d_in[19], *m2gb  = (const float*)d_in[20];
  const float* ad_w  = (const float*)d_in[21], *ad_b  = (const float*)d_in[22];

  // workspace layout (~176 MB)
  char* ws = (char*)d_ws;
  size_t off = 0;
  float*  we   = (float*)(ws + off);  off += (size_t)B_ * E_ * 4;
  float*  xA   = (float*)(ws + off);  off += (size_t)B_ * D_ * 4;
  float*  xB   = (float*)(ws + off);  off += (size_t)B_ * D_ * 4;
  __bf16* hbuf = (__bf16*)(ws + off); off += (size_t)B_ * H_ * 2;
  float*  h2   = (float*)(ws + off);  off += (size_t)B_ * D_ * 4;
  float*  S    = (float*)(ws + off);  off += (size_t)B_ * K_ * 4;
  __bf16* P    = (__bf16*)(ws + off); off += (size_t)B_ * K_ * 2;
  (void)ws_size; (void)in_sizes; (void)n_in; (void)out_size;

  const dim3 blk(256);
  const int MT = B_ / BM;  // 64 row tiles

  // 1) gate + top-2
  gate_topk_kernel<<<B_ / 8, blk, 0, stream>>>(z, gate_w, gate_b, we);
  (void)hipMemsetAsync(xA, 0, (size_t)B_ * D_ * 4, stream);

  // 2) dense experts, weighted combine folded into epilogue
  for (int e = 0; e < E_; ++e) {
    const float* W1 = ew1 + (size_t)e * H_ * D_;
    const float* b1 = eb1 + (size_t)e * H_;
    const float* W2 = ew2 + (size_t)e * D_ * H_;
    const float* b2 = eb2 + (size_t)e * D_;
    wmma_gemm<float, false, EPI_RELU_BF16>
        <<<dim3(H_ / BN, MT), blk, 0, stream>>>(
            z, D_, W1, D_, b1, nullptr, 0, nullptr, nullptr, hbuf, H_, D_, H_);
    wmma_gemm<__bf16, false, EPI_SCALED_ACC_F32>
        <<<dim3(D_ / BN, MT), blk, 0, stream>>>(
            hbuf, H_, W2, H_, b2, we + e, E_, nullptr, nullptr, xA, D_, H_, D_);
  }

  // 3) KV memory: S = z @ keys^T ; softmax ; xA += P @ values
  wmma_gemm<float, false, EPI_STORE_F32>
      <<<dim3(K_ / BN, MT), blk, 0, stream>>>(
          z, D_, keys, D_, nullptr, nullptr, 0, nullptr, nullptr, S, K_, D_, K_);
  row_softmax_bf16<<<B_, blk, 0, stream>>>(S, P, K_);
  wmma_gemm<__bf16, true, EPI_ADD_F32>
      <<<dim3(D_ / BN, MT), blk, 0, stream>>>(
          P, K_, values, D_, nullptr, nullptr, 0, nullptr, nullptr, xA, D_, K_, D_);

  // 4) gMLP block 1: xB = xA + sigmoid(xA@gW^T+gb) * (gelu(xA@f1^T+b1)@f2^T+b2)
  wmma_gemm<float, false, EPI_GELU_BF16>
      <<<dim3(H_ / BN, MT), blk, 0, stream>>>(
          xA, D_, m1f1w, D_, m1f1b, nullptr, 0, nullptr, nullptr, hbuf, H_, D_, H_);
  wmma_gemm<__bf16, false, EPI_STORE_F32>
      <<<dim3(D_ / BN, MT), blk, 0, stream>>>(
          hbuf, H_, m1f2w, H_, m1f2b, nullptr, 0, nullptr, nullptr, h2, D_, H_, D_);
  wmma_gemm<float, false, EPI_GATE_F32>
      <<<dim3(D_ / BN, MT), blk, 0, stream>>>(
          xA, D_, m1gw, D_, m1gb, nullptr, 0, xA, h2, xB, D_, D_, D_);

  // 5) gMLP block 2 (ping-pong back into xA)
  wmma_gemm<float, false, EPI_GELU_BF16>
      <<<dim3(H_ / BN, MT), blk, 0, stream>>>(
          xB, D_, m2f1w, D_, m2f1b, nullptr, 0, nullptr, nullptr, hbuf, H_, D_, H_);
  wmma_gemm<__bf16, false, EPI_STORE_F32>
      <<<dim3(D_ / BN, MT), blk, 0, stream>>>(
          hbuf, H_, m2f2w, H_, m2f2b, nullptr, 0, nullptr, nullptr, h2, D_, H_, D_);
  wmma_gemm<float, false, EPI_GATE_F32>
      <<<dim3(D_ / BN, MT), blk, 0, stream>>>(
          xB, D_, m2gw, D_, m2gb, nullptr, 0, xB, h2, xA, D_, D_, D_);

  // 6) outputs: x then logits (tuple concatenated flat)
  (void)hipMemcpyAsync(d_out, xA, (size_t)B_ * D_ * 4,
                       hipMemcpyDeviceToDevice, stream);
  float* logits = (float*)d_out + (size_t)B_ * D_;
  wmma_gemm<float, false, EPI_STORE_F32>
      <<<dim3((V_ + BN - 1) / BN, MT), blk, 0, stream>>>(
          xA, D_, ad_w, D_, ad_b, nullptr, 0, nullptr, nullptr, logits, V_, D_, V_);
}
